// SeqCLRLoss_45956150067719
// MI455X (gfx1250) — matrix operand order repaired and checked by reading, not compiled, not run on hardware
//
#include <hip/hip_runtime.h>
#include <hip/hip_bf16.h>
#include <math.h>

// ---------------------------------------------------------------------------
// SeqCLR NT-Xent loss, MI455X (gfx1250, wave32, WMMA)
//   views: [128, 32, 512] fp32 each  ->  N = 8192 rows, D = 512
//   loss = mean_i [ log(sum_{j!=i} exp(z_i.z_j/T)) - z_i.z_{(i+4096)%8192}/T ]
// Strategy: L2-normalize rows -> f16 (8 MB, L2-resident), fused WMMA Gram
// matrix with streaming exp-row-sum epilogue (never materialize 8192x8192).
// ---------------------------------------------------------------------------

typedef _Float16 half4  __attribute__((ext_vector_type(4)));
typedef _Float16 half8  __attribute__((ext_vector_type(8)));
typedef _Float16 half16 __attribute__((ext_vector_type(16)));
typedef float    fx8    __attribute__((ext_vector_type(8)));

#define NPV   4096      // instances per view
#define NTOT  8192      // 2 * NPV
#define DDIM  512
#define INVT  10.0f     // 1 / TEMP
#define EPS_N 1e-12f

// ---------------- kernel 0: zero the accumulators -------------------------
__global__ __launch_bounds__(256) void k_zero(float* p, int n) {
    int i = blockIdx.x * 256 + threadIdx.x;
    if (i < n) p[i] = 0.0f;
}

// ---------------- kernel 1: L2 normalize rows -> f16 ----------------------
// one wave (32 lanes) per row; 512 floats = 4 float4 per lane
__global__ __launch_bounds__(256) void k_normalize(const float* __restrict__ v0,
                                                   const float* __restrict__ v1,
                                                   _Float16* __restrict__ z) {
    const int row  = blockIdx.x * 8 + (threadIdx.x >> 5);
    const int lane = threadIdx.x & 31;
    const float* src = (row < NPV) ? (v0 + (size_t)row * DDIM)
                                   : (v1 + (size_t)(row - NPV) * DDIM);
    const float4* s4 = (const float4*)src;

    float4 x[4];
    float ss = 0.0f;
#pragma unroll
    for (int j = 0; j < 4; ++j) {
        x[j] = s4[lane + 32 * j];
        ss += x[j].x * x[j].x + x[j].y * x[j].y + x[j].z * x[j].z + x[j].w * x[j].w;
    }
    // wave32 reduction
#pragma unroll
    for (int m = 1; m < 32; m <<= 1) ss += __shfl_xor(ss, m, 32);

    const float inv = 1.0f / fmaxf(sqrtf(ss), EPS_N);

    half4* dst = (half4*)(z + (size_t)row * DDIM);
#pragma unroll
    for (int j = 0; j < 4; ++j) {
        half4 h;
        h[0] = (_Float16)(x[j].x * inv);
        h[1] = (_Float16)(x[j].y * inv);
        h[2] = (_Float16)(x[j].z * inv);
        h[3] = (_Float16)(x[j].w * inv);
        dst[lane + 32 * j] = h;
    }
}

// ---------------- kernel 2: fused WMMA Gram + exp row-sum -----------------
// grid = (512 row-tiles, 8 column super-groups of 1024 cols); 8 waves/block.
// Each wave: one 16-row tile x 8 column tiles of 16, K = 512 in steps of 32.
__global__ __launch_bounds__(256) void k_gram(const _Float16* __restrict__ z,
                                              float* __restrict__ rowsum,
                                              float* __restrict__ pos) {
    const int R    = blockIdx.x;            // row tile [0,512)
    const int g    = blockIdx.y;            // column super-group [0,8)
    const int w    = threadIdx.x >> 5;      // wave in block [0,8)
    const int lane = threadIdx.x & 31;
    const int hi   = lane >> 4;             // half-wave select
    const int l15  = lane & 15;

    // A fragment source: row = R*16 + (lane&15); K contiguous (matches B):
    //   lane<16: halves 0..15 = K 0..15
    //   lane>=16: halves 0..15 = K 16..31
    const _Float16* arow = z + (size_t)(R * 16 + l15) * DDIM;
    const int akoff = hi << 4;              // +16 for upper half-wave

    float acc[8];
#pragma unroll
    for (int r = 0; r < 8; ++r) acc[r] = 0.0f;

    for (int j = 0; j < 8; ++j) {
        const int colBase = (g * 64 + w * 8 + j) * 16;
        const _Float16* brow = z + (size_t)(colBase + l15) * DDIM;
        const int bkoff = hi << 4;          // B: 16 consecutive K per lane

        fx8 c = {};
        for (int k0 = 0; k0 < DDIM; k0 += 32) {
            // A: two contiguous B128 loads
            half8 alo = *(const half8*)(arow + k0 + akoff);
            half8 ahi = *(const half8*)(arow + k0 + akoff + 8);
            half16 a;
#pragma unroll
            for (int i = 0; i < 8; ++i) { a[i] = alo[i]; a[i + 8] = ahi[i]; }
            // B: one contiguous 32B load (two B128)
            half16 b = *(const half16*)(brow + k0 + bkoff);

            c = __builtin_amdgcn_wmma_f32_16x16x32_f16(
                    false, a, false, b, (short)0, c, false, false);
        }

        // epilogue: C layout -> VGPR r, lanes 0-15: M=r, lanes 16-31: M=r+8;
        // N = lane&15 for both halves.
        const int cg = colBase + l15;
#pragma unroll
        for (int r = 0; r < 8; ++r) {
            const float s  = c[r];
            const int   rg = R * 16 + r + (hi << 3);
            float e = __expf(s * INVT);
            if (rg == cg) e = 0.0f;                       // exclude self-sim
            if (cg == ((rg + NPV) & (NTOT - 1))) pos[rg] = s;  // positive pair
            acc[r] += e;
        }
    }

    // reduce over columns: xor masks 1..8 stay inside each 16-lane half
#pragma unroll
    for (int r = 0; r < 8; ++r) {
#pragma unroll
        for (int m = 1; m < 16; m <<= 1) acc[r] += __shfl_xor(acc[r], m, 32);
    }
    if (l15 == 0) {
#pragma unroll
        for (int r = 0; r < 8; ++r)
            atomicAdd(&rowsum[R * 16 + r + (hi << 3)], acc[r]);
    }
}

// ---------------- kernel 3: final loss mean -------------------------------
__global__ __launch_bounds__(256) void k_finalize(const float* __restrict__ rowsum,
                                                  const float* __restrict__ pos,
                                                  float* __restrict__ out) {
    __shared__ float sdata[256];
    const int t = threadIdx.x;
    float local = 0.0f;
    for (int i = t; i < NTOT; i += 256)
        local += logf(rowsum[i]) - INVT * pos[i];   // log(denom) - log(nom)
    sdata[t] = local;
    __syncthreads();
    for (int s = 128; s > 0; s >>= 1) {
        if (t < s) sdata[t] += sdata[t + s];
        __syncthreads();
    }
    if (t == 0) out[0] = sdata[0] / (float)NTOT;
}

// ---------------------------------------------------------------------------
extern "C" void kernel_launch(void* const* d_in, const int* in_sizes, int n_in,
                              void* d_out, int out_size, void* d_ws, size_t ws_size,
                              hipStream_t stream) {
    const float* v0 = (const float*)d_in[0];
    const float* v1 = (const float*)d_in[1];
    float* out = (float*)d_out;

    char* ws = (char*)d_ws;
    _Float16* z      = (_Float16*)ws;                               // 8 MB
    float*    rowsum = (float*)(ws + (size_t)NTOT * DDIM * 2);      // 32 KB
    float*    pos    = rowsum + NTOT;                               // 32 KB

    // rowsum + pos = 16384 floats, zero both (rowsum needed every launch)
    k_zero<<<64, 256, 0, stream>>>(rowsum, 2 * NTOT);
    k_normalize<<<NTOT / 8, 256, 0, stream>>>(v0, v1, z);
    k_gram<<<dim3(NTOT / 16, 8), 256, 0, stream>>>(z, rowsum, pos);
    k_finalize<<<1, 256, 0, stream>>>(rowsum, pos, out);
}
